// AsyncAttentionMultimodal_1297080124023
// MI455X (gfx1250) — matrix operand order repaired and checked by reading, not compiled, not run on hardware
//
#include <hip/hip_runtime.h>
#include <cstdint>
#include <cstddef>

// ---------------------------------------------------------------------------
// Types for CDNA5 WMMA (wave32, v_wmma_f32_16x16x32_bf16)
// ---------------------------------------------------------------------------
typedef __bf16 bf16;
typedef __attribute__((ext_vector_type(16))) __bf16 v16bf;
typedef __attribute__((ext_vector_type(8)))  float  v8f;
typedef __attribute__((ext_vector_type(4)))  unsigned int u32x4;
typedef __attribute__((ext_vector_type(8)))  int i32x8;
typedef __attribute__((ext_vector_type(4)))  int i32x4;

union Frag16 { u32x4 q[2]; v16bf v; };

// Problem constants (from the reference)
#define NB   32
#define TT   1024
#define NT   (NB * TT)      // 32768 positions
#define AH   352            // padded concat hidden (340 -> 352, mult of 32)
#define HSTR 136            // LDS bf16 row stride (128 + 8) to avoid bank conflicts
#define MAXH 128
#define MAX4H 512

#if __has_builtin(__builtin_amdgcn_tensor_load_to_lds)
#define USE_TDM 1
#else
#define USE_TDM 0
#endif

// ---------------------------------------------------------------------------
// WMMA fragment loaders (ISA 7.12.2 layouts)
// ---------------------------------------------------------------------------
__device__ inline v16bf load_a16x32(const bf16* __restrict__ base, int lda) {
  const int lane = threadIdx.x & 31;
  const int row  = lane & 15;
  const int kb   = (lane & 16) ? 8 : 0;
  Frag16 f;
  f.q[0] = *(const u32x4*)(base + (size_t)row * lda + kb);
  f.q[1] = *(const u32x4*)(base + (size_t)row * lda + kb + 16);
  return f.v;
}

// B: 32x16 bf16 (KxN); source stored row-major as (Nout x K): B[k][n] = W[n][k].
__device__ inline v16bf load_b32x16(const bf16* __restrict__ wNxK, int ldb) {
  const int lane = threadIdx.x & 31;
  const int col  = lane & 15;
  const int kb   = (lane & 16) ? 8 : 0;
  Frag16 f;
  f.q[0] = *(const u32x4*)(wNxK + (size_t)col * ldb + kb);
  f.q[1] = *(const u32x4*)(wNxK + (size_t)col * ldb + kb + 16);
  return f.v;
}

// C/D: 16x16 f32. VGPR r: row = r + ((lane&16)?8:0), col = lane&15.
__device__ inline void store_c16x16(float* __restrict__ C, int ldc, v8f c) {
  const int lane = threadIdx.x & 31;
  const int col  = lane & 15;
  const int rb   = (lane & 16) ? 8 : 0;
#pragma unroll
  for (int r = 0; r < 8; ++r) C[(size_t)(rb + r) * ldc + col] = c[r];
}

// ---------------------------------------------------------------------------
// fp32 -> bf16 convert with column padding (zeros in [cols, colsPad))
// ---------------------------------------------------------------------------
__global__ void __launch_bounds__(256) convert_pad(const float* __restrict__ src,
                                                   bf16* __restrict__ dst,
                                                   int rows, int cols, int colsPad) {
  size_t i   = (size_t)blockIdx.x * 256 + threadIdx.x;
  size_t tot = (size_t)rows * colsPad;
  if (i >= tot) return;
  int    c = (int)(i % colsPad);
  size_t r = i / colsPad;
  dst[i] = (c < cols) ? (bf16)src[r * cols + c] : (bf16)0.f;
}

// attn W1 (64 x 340) remapped to the reordered all_hs column layout:
// dst cols: [0,128)=ling(src 0..127), [128,192)=acoustic(src 148..211),
// [192,320)=image(src 212..339), [320,340)=emotient(src 128..147), rest 0.
__global__ void __launch_bounds__(256) convert_attn_w(const float* __restrict__ src,
                                                      bf16* __restrict__ dst) {
  int i = blockIdx.x * 256 + threadIdx.x;
  if (i >= 64 * AH) return;
  int r = i / AH, c = i % AH;
  int sc;
  if      (c < 128) sc = c;
  else if (c < 192) sc = 148 + (c - 128);
  else if (c < 320) sc = 212 + (c - 192);
  else if (c < 340) sc = 128 + (c - 320);
  else              sc = -1;
  dst[i] = (sc >= 0) ? (bf16)src[r * 340 + sc] : (bf16)0.f;
}

// ---------------------------------------------------------------------------
// WMMA GEMM, 16x16 tile per wave (for Nout not divisible by 32).
// C[M x Nout](f32) = A[M x K](bf16) * B^T, B row-major (Nout x K).
// ---------------------------------------------------------------------------
__global__ void __launch_bounds__(256) gemm_bf16_nt(const bf16* __restrict__ A, int lda,
                                                    const bf16* __restrict__ B, int ldb,
                                                    float* __restrict__ C, int ldc,
                                                    int M, int K) {
  const int wave  = threadIdx.x >> 5;
  const int mTile = blockIdx.y * 8 + wave;
  const int nTile = blockIdx.x;
  if (mTile * 16 >= M) return;
  const bf16* a0 = A + (size_t)mTile * 16 * lda;
  const bf16* b0 = B + (size_t)nTile * 16 * ldb;
  v8f acc = {};
  for (int k = 0; k < K; k += 32) {
    v16bf a = load_a16x32(a0 + k, lda);
    v16bf b = load_b32x16(b0 + k, ldb);
    acc = __builtin_amdgcn_wmma_f32_16x16x32_bf16(false, a, false, b,
                                                  (short)0, acc, false, false);
  }
  store_c16x16(C + (size_t)mTile * 16 * ldc + (size_t)nTile * 16, ldc, acc);
}

// ---------------------------------------------------------------------------
// WMMA GEMM, 2x2 register tile per wave (32x32 of C): 4 frag loads -> 4 WMMA,
// halves per-FLOP memory traffic vs the 16x16 version.
// ---------------------------------------------------------------------------
__global__ void __launch_bounds__(256) gemm_bf16_nt_2x2(const bf16* __restrict__ A, int lda,
                                                        const bf16* __restrict__ B, int ldb,
                                                        float* __restrict__ C, int ldc,
                                                        int M, int K) {
  const int wave  = threadIdx.x >> 5;
  const int mTile = (blockIdx.y * 8 + wave) * 2;
  const int nTile = blockIdx.x * 2;
  if (mTile * 16 >= M) return;
  const bf16* a0 = A + (size_t)mTile * 16 * lda;
  const bf16* a1 = a0 + (size_t)16 * lda;
  const bf16* b0 = B + (size_t)nTile * 16 * ldb;
  const bf16* b1 = b0 + (size_t)16 * ldb;
  v8f c00 = {}, c01 = {}, c10 = {}, c11 = {};
  for (int k = 0; k < K; k += 32) {
    v16bf av0 = load_a16x32(a0 + k, lda);
    v16bf av1 = load_a16x32(a1 + k, lda);
    v16bf bv0 = load_b32x16(b0 + k, ldb);
    v16bf bv1 = load_b32x16(b1 + k, ldb);
    c00 = __builtin_amdgcn_wmma_f32_16x16x32_bf16(false, av0, false, bv0, (short)0, c00, false, false);
    c01 = __builtin_amdgcn_wmma_f32_16x16x32_bf16(false, av0, false, bv1, (short)0, c01, false, false);
    c10 = __builtin_amdgcn_wmma_f32_16x16x32_bf16(false, av1, false, bv0, (short)0, c10, false, false);
    c11 = __builtin_amdgcn_wmma_f32_16x16x32_bf16(false, av1, false, bv1, (short)0, c11, false, false);
  }
  float* Cb = C + (size_t)mTile * 16 * ldc + (size_t)nTile * 16;
  store_c16x16(Cb, ldc, c00);
  store_c16x16(Cb + 16, ldc, c01);
  store_c16x16(Cb + (size_t)16 * ldc, ldc, c10);
  store_c16x16(Cb + (size_t)16 * ldc + 16, ldc, c11);
}

// ---------------------------------------------------------------------------
// Tensor Data Mover: stage one timestep's G tile [32 x 4H] f32 into LDS.
// D# per ISA 8.3/8.4: 2D tile, data_size=4B, tile_dim0=FH, tile_dim1=32,
// tensor_dim0_stride = T*FH elements (batch-row stride in G).
// This toolchain exposes the 6-arg builtin:
//   (u32x4 g0, i32x8 g1, i32x4 g2, i32x4 g3, i32x8 gx, i32 cpol)
// ---------------------------------------------------------------------------
#if USE_TDM
__device__ inline void tdm_load_G(const float* gsrc, unsigned ldsOff, int FH) {
  unsigned long long ga = (unsigned long long)(size_t)gsrc;
  u32x4 g0;
  g0[0] = 1u;                                            // count=1 (valid D#)
  g0[1] = ldsOff;                                        // lds_addr (bytes)
  g0[2] = (unsigned)(ga & 0xffffffffu);                  // global_addr[31:0]
  g0[3] = (unsigned)((ga >> 32) & 0x01ffffffu) | (2u << 30); // addr[56:32] | type=2
  i32x8 g1;
  g1[0] = 0x00020000;                 // wg_mask=0, data_size=2 (4 bytes)
  g1[1] = FH << 16;                   // tensor_dim0[15:0] (atomic_barrier_addr=0)
  g1[2] = NB << 16;                   // tensor_dim0 hi=0 | tensor_dim1[15:0]=32
  g1[3] = FH << 16;                   // tensor_dim1 hi=0 | tile_dim0=FH
  g1[4] = NB;                         // tile_dim1=32, tile_dim2=0
  g1[5] = TT * FH;                    // tensor_dim0_stride[31:0] (elements)
  g1[6] = 0;                          // stride hi / dim1_stride
  g1[7] = 0;
  i32x4 z4 = {0, 0, 0, 0};            // groups 2/3 unused (2D tensor)
  i32x8 z8 = {0, 0, 0, 0, 0, 0, 0, 0};
  __builtin_amdgcn_tensor_load_to_lds(g0, g1, z4, z4, z8, 0);
}
#endif

// ---------------------------------------------------------------------------
// Sequential LSTM scan: one workgroup (8 wave32) per modality.
// Per step: WMMA h@Whh^T out of LDS; TDM double-buffers next step's G tile;
// VALU applies sigmoid/tanh; masked h -> concatenated all_hs (bf16).
// ---------------------------------------------------------------------------
struct ScanParams {
  const float* G[4];      // [NT][4H] gate pre-activations (row = n*T + t)
  const bf16*  Whh[4];    // [4H][Hp] bf16 (K zero-padded)
  const float* bih[4];
  const float* bhh[4];
  const int*   seqlen;    // [N]
  bf16*        all_hs;    // [NT][AH] bf16, reordered columns
  int H[4], Hp[4], FH[4], colOff[4];
};

__device__ inline float sigf(float x) { return 1.f / (1.f + __expf(-x)); }

__global__ void __launch_bounds__(256) lstm_scan(ScanParams p) {
  __shared__ bf16  hbf[NB * HSTR];     // h state, bf16, padded stride
  __shared__ float cbuf[NB * MAXH];    // c state, f32
  __shared__ float gbuf[NB * MAX4H];   // recurrent gate contributions, f32
#if USE_TDM
  __shared__ float Gst[2][NB * MAX4H]; // double-buffered G tile staging
#endif

  const int mod = blockIdx.x;
  const int H = p.H[mod], Hp = p.Hp[mod], FH = p.FH[mod], co = p.colOff[mod];
  const int tid = threadIdx.x, wave = tid >> 5;

  for (int i = tid; i < NB * HSTR; i += 256) hbf[i]  = (bf16)0.f;
  for (int i = tid; i < NB * MAXH; i += 256) cbuf[i] = 0.f;
  __syncthreads();

  const int nTiles   = FH >> 4;
  const int totTiles = nTiles * 2;          // 2 M-tiles (batch 32) x nTiles
  const float* __restrict__ Gt  = p.G[mod];
  const bf16*  __restrict__ Whh = p.Whh[mod];
  const float* __restrict__ bih = p.bih[mod];
  const float* __restrict__ bhh = p.bhh[mod];

#if USE_TDM
  unsigned gstOff[2] = { (unsigned)(size_t)(&Gst[0][0]),
                         (unsigned)(size_t)(&Gst[1][0]) };
  if (wave == 0) tdm_load_G(Gt, gstOff[0], FH);   // prologue: stage t=0
#endif

  for (int t = 0; t < TT; ++t) {
#if USE_TDM
    if (wave == 0 && t + 1 < TT)                  // stage t+1 while computing t
      tdm_load_G(Gt + (size_t)(t + 1) * FH, gstOff[(t + 1) & 1], FH);
#endif

    // ---- phase 1: gbuf = h @ Whh^T (WMMA out of LDS) ----
    for (int tile = wave; tile < totTiles; tile += 8) {
      const int mT = tile & 1, nT = tile >> 1;
      const bf16* a0 = hbf + mT * 16 * HSTR;
      const bf16* b0 = Whh + (size_t)nT * 16 * Hp;
      v8f acc = {};
      for (int k = 0; k < Hp; k += 32) {
        v16bf a = load_a16x32(a0 + k, HSTR);
        v16bf b = load_b32x16(b0 + k, Hp);
        acc = __builtin_amdgcn_wmma_f32_16x16x32_bf16(false, a, false, b,
                                                      (short)0, acc, false, false);
      }
      store_c16x16(gbuf + (size_t)mT * 16 * FH + nT * 16, FH, acc);
    }

#if USE_TDM
    if (wave == 0) {                 // in-order TENSORcnt: <=1 pending => t done
      if (t + 1 < TT) __builtin_amdgcn_s_wait_tensorcnt(1);
      else            __builtin_amdgcn_s_wait_tensorcnt(0);
    }
#endif
    __syncthreads();                 // publishes gbuf + staged G tile

    // ---- phase 2: nonlinearities + state update ----
#if USE_TDM
    const float* __restrict__ Gcur = &Gst[t & 1][0];
#endif
    for (int e = tid; e < NB * H; e += 256) {
      const int m = e / H, j = e - m * H;
#if USE_TDM
      const int gr = m * FH;
      const float iv = gbuf[gr + j]         + Gcur[gr + j]         + bih[j]         + bhh[j];
      const float fv = gbuf[gr + H + j]     + Gcur[gr + H + j]     + bih[H + j]     + bhh[H + j];
      const float gv = gbuf[gr + 2 * H + j] + Gcur[gr + 2 * H + j] + bih[2 * H + j] + bhh[2 * H + j];
      const float ov = gbuf[gr + 3 * H + j] + Gcur[gr + 3 * H + j] + bih[3 * H + j] + bhh[3 * H + j];
#else
      const size_t grow = ((size_t)m * TT + t) * FH;
      const float iv = gbuf[m * FH + j]         + Gt[grow + j]         + bih[j]         + bhh[j];
      const float fv = gbuf[m * FH + H + j]     + Gt[grow + H + j]     + bih[H + j]     + bhh[H + j];
      const float gv = gbuf[m * FH + 2 * H + j] + Gt[grow + 2 * H + j] + bih[2 * H + j] + bhh[2 * H + j];
      const float ov = gbuf[m * FH + 3 * H + j] + Gt[grow + 3 * H + j] + bih[3 * H + j] + bhh[3 * H + j];
#endif
      const float c  = sigf(fv) * cbuf[m * MAXH + j] + sigf(iv) * tanhf(gv);
      const float h  = sigf(ov) * tanhf(c);
      cbuf[m * MAXH + j] = c;
      hbf[m * HSTR + j]  = (bf16)h;
      const float hm = (t < p.seqlen[m]) ? h : 0.f;
      p.all_hs[((size_t)m * TT + t) * AH + co + j] = (bf16)hm;
    }
    __syncthreads();
  }
}

// ---------------------------------------------------------------------------
// Final head: per position combine attn hidden + per-mod rating hidden.
// ---------------------------------------------------------------------------
struct HeadParams {
  const float* attnH;     // [NT][64]  = all_hs @ W1a^T   (no bias yet)
  const float* Um[4];     // [NT][16]  = hs_mod @ W1m^T   (no bias yet)
  const float* b1m[4]; const float* W2m[4]; const float* b2m[4];
  const float* b1a; const float* W2a; const float* b2a;   // 64, 4x64, 4
  const float* Wf; const float* bf_;                      // 1x4, 1
  const float* mask;                                      // [NT] lstm_masks
  float* out;                                             // [NT]
};

__global__ void __launch_bounds__(256) head_finalize(HeadParams hp) {
  const int i = blockIdx.x * 256 + threadIdx.x;
  if (i >= NT) return;
  float rating[4];
#pragma unroll
  for (int m = 0; m < 4; ++m) {
    float s = hp.b2m[m][0];
    const float* u = hp.Um[m] + (size_t)i * 16;
#pragma unroll
    for (int k = 0; k < 16; ++k) s += hp.W2m[m][k] * (u[k] + hp.b1m[m][k]);
    rating[m] = s;
  }
  const float* ah = hp.attnH + (size_t)i * 64;
  float o = hp.bf_[0];
#pragma unroll
  for (int m = 0; m < 4; ++m) {
    float a = hp.b2a[m];
    for (int k = 0; k < 64; ++k) a += hp.W2a[m * 64 + k] * (ah[k] + hp.b1a[k]);
    o += hp.Wf[m] * (rating[m] * a);
  }
  hp.out[i] = o * hp.mask[i];
}

// ---------------------------------------------------------------------------
// Host launcher
// ---------------------------------------------------------------------------
extern "C" void kernel_launch(void* const* d_in, const int* in_sizes, int n_in,
                              void* d_out, int out_size, void* d_ws, size_t ws_size,
                              hipStream_t stream) {
  (void)in_sizes; (void)n_in; (void)out_size; (void)ws_size;

  // Modality order: linguistic, emotient, acoustic, image
  static const int Dm[4] = {300, 40, 88, 1000};
  static const int Dp[4] = {320, 64, 96, 1024};
  static const int Hm[4] = {128, 20, 64, 128};
  static const int Hp[4] = {128, 32, 64, 128};
  static const int FH[4] = {512, 80, 256, 512};
  static const int CO[4] = {0, 320, 128, 192};   // 16B-aligned all_hs columns

  const float* x_in[4]   = {(const float*)d_in[0], (const float*)d_in[1],
                            (const float*)d_in[2], (const float*)d_in[3]};
  const int*   seqlen    = (const int*)d_in[4];
  const float* masks     = (const float*)d_in[5];
  const float *Wih[4], *Whh[4], *bih[4], *bhh[4];
  for (int m = 0; m < 4; ++m) {
    Wih[m] = (const float*)d_in[6 + 4 * m + 0];
    Whh[m] = (const float*)d_in[6 + 4 * m + 1];
    bih[m] = (const float*)d_in[6 + 4 * m + 2];
    bhh[m] = (const float*)d_in[6 + 4 * m + 3];
  }
  const float *W1m[4], *b1m[4], *W2m[4], *b2m[4];
  for (int m = 0; m < 4; ++m) {
    W1m[m] = (const float*)d_in[22 + 4 * m + 0];
    b1m[m] = (const float*)d_in[22 + 4 * m + 1];
    W2m[m] = (const float*)d_in[22 + 4 * m + 2];
    b2m[m] = (const float*)d_in[22 + 4 * m + 3];
  }
  const float* W1a = (const float*)d_in[38];
  const float* b1a = (const float*)d_in[39];
  const float* W2a = (const float*)d_in[40];
  const float* b2a = (const float*)d_in[41];
  const float* Wf  = (const float*)d_in[42];
  const float* bf_ = (const float*)d_in[43];

  // ---- workspace layout ----
  char* ws = (char*)d_ws;
  size_t cur = 0;
  auto alloc = [&](size_t bytes) {
    void* p = ws + cur;
    cur = (cur + bytes + 255) & ~(size_t)255;
    return p;
  };
  bf16 *xb[4], *Wihb[4], *Whhb[4], *W1mb[4];
  float *G[4], *Um[4];
  for (int m = 0; m < 4; ++m) xb[m]   = (bf16*)alloc((size_t)NT * Dp[m] * 2);
  for (int m = 0; m < 4; ++m) Wihb[m] = (bf16*)alloc((size_t)FH[m] * Dp[m] * 2);
  for (int m = 0; m < 4; ++m) Whhb[m] = (bf16*)alloc((size_t)FH[m] * Hp[m] * 2);
  for (int m = 0; m < 4; ++m) W1mb[m] = (bf16*)alloc((size_t)16 * Hp[m] * 2);
  bf16* W1ab   = (bf16*)alloc((size_t)64 * AH * 2);
  for (int m = 0; m < 4; ++m) G[m]    = (float*)alloc((size_t)NT * FH[m] * 4);
  bf16* all_hs = (bf16*)alloc((size_t)NT * AH * 2);
  float* attnH = (float*)alloc((size_t)NT * 64 * 4);
  for (int m = 0; m < 4; ++m) Um[m]   = (float*)alloc((size_t)NT * 16 * 4);

  auto cvt = [&](const float* src, bf16* dst, int rows, int cols, int colsPad) {
    size_t tot = (size_t)rows * colsPad;
    int blocks = (int)((tot + 255) / 256);
    convert_pad<<<blocks, 256, 0, stream>>>(src, dst, rows, cols, colsPad);
  };

  // 1) convert inputs + weights to bf16 (zero K-padding)
  for (int m = 0; m < 4; ++m) cvt(x_in[m], xb[m],  NT,    Dm[m], Dp[m]);
  for (int m = 0; m < 4; ++m) cvt(Wih[m],  Wihb[m], FH[m], Dm[m], Dp[m]);
  for (int m = 0; m < 4; ++m) cvt(Whh[m],  Whhb[m], FH[m], Hm[m], Hp[m]);
  for (int m = 0; m < 4; ++m) cvt(W1m[m],  W1mb[m], 16,    Hm[m], Hp[m]);
  convert_attn_w<<<(64 * AH + 255) / 256, 256, 0, stream>>>(W1a, W1ab);
  (void)hipMemsetAsync(all_hs, 0, (size_t)NT * AH * 2, stream);  // zero pad/gap cols

  // 2) input projections: G[m] = x @ Wih^T  (~47 GFLOP bulk, 2x2-tiled WMMA)
  for (int m = 0; m < 4; ++m) {
    if ((FH[m] & 31) == 0) {
      dim3 grid(FH[m] / 32, NT / (32 * 8));
      gemm_bf16_nt_2x2<<<grid, 256, 0, stream>>>(xb[m], Dp[m], Wihb[m], Dp[m],
                                                 G[m], FH[m], NT, Dp[m]);
    } else {
      dim3 grid(FH[m] / 16, NT / (16 * 8));
      gemm_bf16_nt<<<grid, 256, 0, stream>>>(xb[m], Dp[m], Wihb[m], Dp[m],
                                             G[m], FH[m], NT, Dp[m]);
    }
  }

  // 3) sequential LSTM scans (one WGP-resident block per modality; TDM-fed)
  ScanParams sp;
  for (int m = 0; m < 4; ++m) {
    sp.G[m] = G[m]; sp.Whh[m] = Whhb[m]; sp.bih[m] = bih[m]; sp.bhh[m] = bhh[m];
    sp.H[m] = Hm[m]; sp.Hp[m] = Hp[m]; sp.FH[m] = FH[m]; sp.colOff[m] = CO[m];
  }
  sp.seqlen = seqlen; sp.all_hs = all_hs;
  lstm_scan<<<4, 256, 0, stream>>>(sp);

  // 4) head GEMMs (WMMA): attn hidden + per-mod rating hidden
  {
    dim3 grid(64 / 32, NT / (32 * 8));
    gemm_bf16_nt_2x2<<<grid, 256, 0, stream>>>(all_hs, AH, W1ab, AH, attnH, 64, NT, AH);
  }
  for (int m = 0; m < 4; ++m) {
    dim3 grid(1, NT / (16 * 8));
    gemm_bf16_nt<<<grid, 256, 0, stream>>>(all_hs + CO[m], AH, W1mb[m], Hp[m],
                                           Um[m], 16, NT, Hp[m]);
  }

  // 5) finalize: ratings * attn -> final linear -> mask
  HeadParams hp;
  hp.attnH = attnH;
  for (int m = 0; m < 4; ++m) {
    hp.Um[m] = Um[m]; hp.b1m[m] = b1m[m]; hp.W2m[m] = W2m[m]; hp.b2m[m] = b2m[m];
  }
  hp.b1a = b1a; hp.W2a = W2a; hp.b2a = b2a; hp.Wf = Wf; hp.bf_ = bf_;
  hp.mask = masks; hp.out = (float*)d_out;
  head_finalize<<<NT / 256, 256, 0, stream>>>(hp);
}
